// PCFG_34780645163010
// MI455X (gfx1250) — matrix operand order repaired and checked by reading, not compiled, not run on hardware
//
#include <hip/hip_runtime.h>
#include <hip/hip_bf16.h>
#include <math.h>

#define NTC 48
#define DC 64
#define VC 8000
#define BATCH 16
#define LSENT 16
#define DH 128                 // 2*D (emission MLP width)
#define NROWS (NTC * VC)       // 384000

typedef _Float16 half_t;
typedef half_t v8h  __attribute__((ext_vector_type(8)));
typedef half_t v16h __attribute__((ext_vector_type(16)));
typedef float  v8f  __attribute__((ext_vector_type(8)));

// ---------------------------------------------------------------------------
// Block reductions (blockDim.x == 256)
// ---------------------------------------------------------------------------
__device__ float blockReduceMax(float v, float* red) {
    int tid = threadIdx.x;
    red[tid] = v; __syncthreads();
    for (int s = 128; s > 0; s >>= 1) {
        if (tid < s) red[tid] = fmaxf(red[tid], red[tid + s]);
        __syncthreads();
    }
    float r = red[0]; __syncthreads();
    return r;
}

__device__ float blockReduceSum(float v, float* red) {
    int tid = threadIdx.x;
    red[tid] = v; __syncthreads();
    for (int s = 128; s > 0; s >>= 1) {
        if (tid < s) red[tid] += red[tid + s];
        __syncthreads();
    }
    float r = red[0]; __syncthreads();
    return r;
}

// ---------------------------------------------------------------------------
// Serial 4-layer MLP (din,dh <= 64) used for the tiny root/split networks
// ---------------------------------------------------------------------------
__device__ void mlp4_dev(const float* x, int din, int dh, int dout,
                         const float* Wi,  const float* bi,
                         const float* W11, const float* b11,
                         const float* W12, const float* b12,
                         const float* W21, const float* b21,
                         const float* W22, const float* b22,
                         const float* Wo,  const float* bo, float* out) {
    float h[64], t[64], u[64];
    for (int j = 0; j < dh; ++j) {
        float s = bi[j]; const float* w = Wi + j * din;
        for (int k = 0; k < din; ++k) s += w[k] * x[k];
        h[j] = s;
    }
    for (int r = 0; r < 2; ++r) {
        const float* W1 = r ? W21 : W11; const float* B1 = r ? b21 : b11;
        const float* W2 = r ? W22 : W12; const float* B2 = r ? b22 : b12;
        for (int j = 0; j < dh; ++j) {
            float s = B1[j]; const float* w = W1 + j * dh;
            for (int k = 0; k < dh; ++k) s += w[k] * h[k];
            t[j] = fmaxf(s, 0.f);
        }
        for (int j = 0; j < dh; ++j) {
            float s = B2[j]; const float* w = W2 + j * dh;
            for (int k = 0; k < dh; ++k) s += w[k] * t[k];
            u[j] = fmaxf(s, 0.f) + h[j];
        }
        for (int j = 0; j < dh; ++j) h[j] = u[j];
    }
    for (int j = 0; j < dout; ++j) {
        float s = bo[j]; const float* w = Wo + j * dh;
        for (int k = 0; k < dh; ++k) s += w[k] * h[k];
        out[j] = s;
    }
}

// ---------------------------------------------------------------------------
// K0: fp32 -> f16 conversion of embeddings + emission weights
// ---------------------------------------------------------------------------
__global__ void convert_f16(const float* word, const float* nont,
                            const float* w0, const float* w1, const float* w2,
                            const float* w3, const float* w4, const float* wout,
                            half_t* wordH, half_t* nontH, half_t* emitWH,
                            half_t* outWH) {
    int i = blockIdx.x * 256 + threadIdx.x;
    if (i < 512000) {
        wordH[i] = (half_t)word[i];
    } else if (i < 515072) {
        nontH[i - 512000] = (half_t)nont[i - 512000];
    } else if (i < 515072 + 5 * 16384) {
        int j = i - 515072;
        int l = j >> 14, k = j & 16383;
        const float* src = (l == 0) ? w0 : (l == 1) ? w1 : (l == 2) ? w2
                          : (l == 3) ? w3 : w4;
        emitWH[j] = (half_t)src[k];
    } else if (i < 515072 + 5 * 16384 + 128) {
        int k = i - (515072 + 5 * 16384);
        outWH[k] = (half_t)wout[k];
    }
}

// ---------------------------------------------------------------------------
// K1: root MLP (+log_softmax) and split MLP (+log_softmax over 2)
// ---------------------------------------------------------------------------
__global__ void small_mlps(
    const float* nont, const float* root_emb,
    const float* rWi, const float* rbi, const float* rW11, const float* rb11,
    const float* rW12, const float* rb12, const float* rW21, const float* rb21,
    const float* rW22, const float* rb22, const float* rWo, const float* rbo,
    const float* sWi, const float* sbi, const float* sW11, const float* sb11,
    const float* sW12, const float* sb12, const float* sW21, const float* sb21,
    const float* sW22, const float* sb22, const float* sWo, const float* sbo,
    float* root_scores, float* split0, float* split1) {
    int tid = threadIdx.x;
    if (tid == 0) {
        float o[48];
        mlp4_dev(root_emb, 64, 64, 48, rWi, rbi, rW11, rb11, rW12, rb12,
                 rW21, rb21, rW22, rb22, rWo, rbo, o);
        float m = -INFINITY;
        for (int n = 0; n < 48; ++n) m = fmaxf(m, o[n]);
        float s = 0.f;
        for (int n = 0; n < 48; ++n) s += expf(o[n] - m);
        float l = m + logf(s);
        for (int n = 0; n < 48; ++n) root_scores[n] = o[n] - l;
    }
    if (tid < 48) {
        float o[2];
        mlp4_dev(nont + tid * 64, 64, 64, 2, sWi, sbi, sW11, sb11, sW12, sb12,
                 sW21, sb21, sW22, sb22, sWo, sbo, o);
        float m = fmaxf(o[0], o[1]);
        float l = m + logf(expf(o[0] - m) + expf(o[1] - m));
        split0[tid] = o[0] - l;
        split1[tid] = o[1] - l;
    }
}

// ---------------------------------------------------------------------------
// K2: rule linear + row log_softmax folded into expG / mG
//   expG[n,j] = exp(rule_row[j] - max_row);  mG[n] = split0[n] - log(sum exp)
// ---------------------------------------------------------------------------
__global__ void rule_kernel(const float* nont, const float* ruleW,
                            const float* ruleB, const float* split0,
                            float* expG, float* mG) {
    int n = blockIdx.x, tid = threadIdx.x;
    __shared__ float x[64];
    __shared__ float rowv[2304];
    __shared__ float red[256];
    if (tid < 64) x[tid] = nont[n * 64 + tid];
    __syncthreads();
    float lmax = -INFINITY;
    for (int j = tid; j < 2304; j += 256) {
        const float* w = ruleW + j * 64;
        float s = ruleB[j];
        for (int k = 0; k < 64; ++k) s += x[k] * w[k];
        rowv[j] = s;
        lmax = fmaxf(lmax, s);
    }
    float m = blockReduceMax(lmax, red);
    float ls = 0.f;
    for (int j = tid; j < 2304; j += 256) {
        float e = expf(rowv[j] - m);
        expG[n * 2304 + j] = e;
        ls += e;
    }
    float S = blockReduceSum(ls, red);
    if (tid == 0) mG[n] = split0[n] - logf(S);
}

// ---------------------------------------------------------------------------
// K3: emission MLP via WMMA f16 (f32 accumulate).
//   128 threads = 4 wave32s; 64 rows/block; all activations in LDS (f16).
// ---------------------------------------------------------------------------
__device__ __forceinline__ v16h loadAfrag(const half_t* __restrict__ src,
                                          int row, int kb, int hi) {
    // A layout (ISA 7.12.2): lane<16 holds row, K = {kb+hi*8..+7, kb+16+hi*8..+7}
    const half_t* p = src + row * DH + kb + hi * 8;
    v8h a0 = *(const v8h*)p;
    v8h a1 = *(const v8h*)(p + 16);
    return __builtin_shufflevector(a0, a1, 0, 1, 2, 3, 4, 5, 6, 7,
                                   8, 9, 10, 11, 12, 13, 14, 15);
}

__device__ __forceinline__ v16h loadBfrag(const half_t* __restrict__ wb,
                                          int col, int kb, int hi) {
    // B layout: lane<16 holds column col, K = {kb+hi*8..+7, kb+16+hi*8..+7}
    const half_t* p = wb + col * DH + kb + hi * 8;
    v8h b0 = *(const v8h*)p;
    v8h b1 = *(const v8h*)(p + 16);
    return __builtin_shufflevector(b0, b1, 0, 1, 2, 3, 4, 5, 6, 7,
                                   8, 9, 10, 11, 12, 13, 14, 15);
}

// MODE: 0 = linear, 1 = relu, 2 = relu + residual(resid)
template <int MODE>
__device__ __forceinline__ void gemm_layer(const half_t* __restrict__ src,
                                           half_t* __restrict__ dst,
                                           const half_t* __restrict__ resid,
                                           const half_t* __restrict__ wb,
                                           const float* __restrict__ bias,
                                           int wrow, int sub, int hi) {
    int arow = wrow + sub;
    v16h A0 = loadAfrag(src, arow, 0, hi);
    v16h A1 = loadAfrag(src, arow, 32, hi);
    v16h A2 = loadAfrag(src, arow, 64, hi);
    v16h A3 = loadAfrag(src, arow, 96, hi);
#pragma unroll
    for (int nb = 0; nb < 8; ++nb) {
        int col = nb * 16 + sub;
        float bc = bias[col];
        v8f acc = {};
        acc = __builtin_amdgcn_wmma_f32_16x16x32_f16(
            false, A0, false, loadBfrag(wb, col, 0, hi), (short)0, acc, false, false);
        acc = __builtin_amdgcn_wmma_f32_16x16x32_f16(
            false, A1, false, loadBfrag(wb, col, 32, hi), (short)0, acc, false, false);
        acc = __builtin_amdgcn_wmma_f32_16x16x32_f16(
            false, A2, false, loadBfrag(wb, col, 64, hi), (short)0, acc, false, false);
        acc = __builtin_amdgcn_wmma_f32_16x16x32_f16(
            false, A3, false, loadBfrag(wb, col, 96, hi), (short)0, acc, false, false);
#pragma unroll
        for (int i = 0; i < 8; ++i) {
            int row = wrow + hi * 8 + i;
            float y = acc[i] + bc;
            if (MODE >= 1) y = fmaxf(y, 0.f);
            if (MODE == 2) y += (float)resid[row * DH + col];
            dst[row * DH + col] = (half_t)y;
        }
    }
}

__device__ __forceinline__ void stage_weights(half_t* __restrict__ wbuf,
                                              const half_t* __restrict__ emitWH,
                                              int l, int tid) {
    const v8h* wsrc = (const v8h*)(emitWH + l * DH * DH);
    v8h* wdst = (v8h*)wbuf;
#pragma unroll
    for (int i = 0; i < (DH * DH) / 8 / 128; ++i)
        wdst[tid + i * 128] = wsrc[tid + i * 128];
    if (l < 4)
        __builtin_prefetch(emitWH + (l + 1) * DH * DH + tid * 128, 0, 1);
}

__global__ void __launch_bounds__(128) emit_kernel(
    const half_t* __restrict__ wordH, const half_t* __restrict__ nontH,
    const half_t* __restrict__ emitWH, const half_t* __restrict__ outWH,
    const float* __restrict__ bInp, const float* __restrict__ br11,
    const float* __restrict__ br12, const float* __restrict__ br21,
    const float* __restrict__ br22, const float* __restrict__ outB,
    float* __restrict__ scores) {
    __shared__ half_t wbuf[DH * DH];     // 32 KB  (current layer weights)
    __shared__ half_t actA[64 * DH];     // 16 KB
    __shared__ half_t actB[64 * DH];     // 16 KB
    int tid = threadIdx.x;
    int lane = tid & 31, wave = tid >> 5;
    int sub = lane & 15, hi = lane >> 4;
    int rowBase = blockIdx.x * 64;
    int wrow = wave * 16;

    // Stage inputs: row r=(n,v) is concat(word_emb[v], nont_emb[n]) -> actB
#pragma unroll
    for (int c0 = 0; c0 < 8; ++c0) {
        int c = tid + c0 * 128;
        int rl = c >> 4;
        int k0 = (c & 15) * 8;
        int r = rowBase + rl;
        int n = r / VC;
        int v = r - n * VC;
        const v8h* srcp = (k0 < 64) ? (const v8h*)(wordH + v * 64 + k0)
                                    : (const v8h*)(nontH + n * 64 + (k0 - 64));
        *(v8h*)(actB + rl * DH + k0) = *srcp;
    }
    __syncthreads();

    stage_weights(wbuf, emitWH, 0, tid);
    __syncthreads();
    gemm_layer<0>(actB, actA, nullptr, wbuf, bInp, wrow, sub, hi);   // inp
    __syncthreads();
    stage_weights(wbuf, emitWH, 1, tid);
    __syncthreads();
    gemm_layer<1>(actA, actB, nullptr, wbuf, br11, wrow, sub, hi);   // res1.lin1
    __syncthreads();
    stage_weights(wbuf, emitWH, 2, tid);
    __syncthreads();
    gemm_layer<2>(actB, actA, actA, wbuf, br12, wrow, sub, hi);      // res1.lin2
    __syncthreads();
    stage_weights(wbuf, emitWH, 3, tid);
    __syncthreads();
    gemm_layer<1>(actA, actB, nullptr, wbuf, br21, wrow, sub, hi);   // res2.lin1
    __syncthreads();
    stage_weights(wbuf, emitWH, 4, tid);
    __syncthreads();
    gemm_layer<2>(actB, actA, actA, wbuf, br22, wrow, sub, hi);      // res2.lin2
    __syncthreads();

    // out layer (dout=1): half-row dot per lane, pairwise shuffle combine
    {
        int row = wrow + sub;
        int r = rowBase + row;
        const half_t* hv = actA + row * DH + hi * 64;
        const half_t* wv = outWH + hi * 64;
        float s = 0.f;
#pragma unroll
        for (int k = 0; k < 64; ++k) s += (float)hv[k] * (float)wv[k];
        s += __shfl_xor(s, 16);
        if (hi == 0) scores[r] = s + outB[0];
    }
}

// ---------------------------------------------------------------------------
// K4: per-NT logsumexp over the vocab (softmax denominator)
// ---------------------------------------------------------------------------
__global__ void lse_kernel(const float* scores, float* lse) {
    int n = blockIdx.x, tid = threadIdx.x;
    __shared__ float red[256];
    float m = -INFINITY;
    for (int v = tid; v < VC; v += 256) m = fmaxf(m, scores[n * VC + v]);
    m = blockReduceMax(m, red);
    float s = 0.f;
    for (int v = tid; v < VC; v += 256) s += expf(scores[n * VC + v] - m);
    s = blockReduceSum(s, red);
    if (tid == 0) lse[n] = m + logf(s);
}

// ---------------------------------------------------------------------------
// K5: lexical scores -> beta[1]
// ---------------------------------------------------------------------------
__global__ void lex_kernel(const int* words, const float* scores,
                           const float* lse, const float* split1, float* beta) {
    int idx = blockIdx.x * blockDim.x + threadIdx.x;
    if (idx >= BATCH * LSENT * NTC) return;
    int n = idx % NTC;
    int bl = idx / NTC;
    int l = bl % LSENT;
    int b = bl / LSENT;
    int wv = words[b * LSENT + l];
    beta[(((size_t)1 * BATCH + b) * LSENT + l) * NTC + n] =
        scores[n * VC + wv] - lse[n] + split1[n];
}

// ---------------------------------------------------------------------------
// K6: one CKY width step. Block = one (b,p) chart cell.
// ---------------------------------------------------------------------------
__global__ void cky_kernel(int w, const float* __restrict__ expG,
                           const float* __restrict__ mG, float* beta) {
    int P = 17 - w;
    int b = blockIdx.x / P, p = blockIdx.x % P;
    int tid = threadIdx.x;
    int lane = tid & 31;
    __shared__ float Ls[15 * 48];
    __shared__ float Rs[15 * 48];
    __shared__ float red[256];
    __shared__ float tacc[48];

    int cnt = (w - 1) * 48;
    for (int t = tid; t < cnt; t += 256) {
        int k = t / 48 + 1;
        int n = t - (k - 1) * 48;
        Ls[t] = beta[(((size_t)k * BATCH + b) * LSENT + p) * NTC + n];
        Rs[t] = beta[(((size_t)(w - k) * BATCH + b) * LSENT + (p + k)) * NTC + n];
    }
    if (tid < 48) tacc[tid] = 0.f;
    __syncthreads();

    // Y[j] = logsumexp_k(L_k[n1] + R_k[n2]); 2304 = 9 * 256 exactly
    float yv[9];
    float lmax = -INFINITY;
#pragma unroll
    for (int i = 0; i < 9; ++i) {
        int j = tid + 256 * i;
        int n1 = j / 48, n2 = j - n1 * 48;
        float m = -INFINITY;
        for (int k = 0; k < w - 1; ++k)
            m = fmaxf(m, Ls[k * 48 + n1] + Rs[k * 48 + n2]);
        float s = 0.f;
        for (int k = 0; k < w - 1; ++k)
            s += expf(Ls[k * 48 + n1] + Rs[k * 48 + n2] - m);
        float y = m + logf(s);
        yv[i] = y;
        lmax = fmaxf(lmax, y);
    }
    float mY = blockReduceMax(lmax, red);
    float ev[9];
#pragma unroll
    for (int i = 0; i < 9; ++i) ev[i] = expf(yv[i] - mY);

    // t[a] = sum_j ev_j * expG[a, j]  (wave shuffle reduce + ds_add_f32)
    for (int ab = 0; ab < 48; ab += 8) {
        float pa[8] = {0, 0, 0, 0, 0, 0, 0, 0};
        for (int i = 0; i < 9; ++i) {
            int j = tid + 256 * i;
            float e = ev[i];
#pragma unroll
            for (int q = 0; q < 8; ++q) pa[q] += e * expG[(ab + q) * 2304 + j];
        }
#pragma unroll
        for (int q = 0; q < 8; ++q) {
            float v = pa[q];
            for (int off = 16; off > 0; off >>= 1) v += __shfl_xor(v, off);
            if (lane == 0) atomicAdd(&tacc[ab + q], v);
        }
    }
    __syncthreads();
    if (tid < 48) {
        beta[(((size_t)w * BATCH + b) * LSENT + p) * NTC + tid] =
            logf(tacc[tid]) + mY + mG[tid];
    }
}

// ---------------------------------------------------------------------------
// K7: -logZ per batch element
// ---------------------------------------------------------------------------
__global__ void final_kernel(const float* root, const float* beta, float* out) {
    int b = threadIdx.x;
    if (b >= BATCH) return;
    const float* bp = beta + (((size_t)16 * BATCH + b) * LSENT + 0) * NTC;
    float m = -INFINITY;
    for (int n = 0; n < NTC; ++n) m = fmaxf(m, root[n] + bp[n]);
    float s = 0.f;
    for (int n = 0; n < NTC; ++n) s += expf(root[n] + bp[n] - m);
    out[b] = -(m + logf(s));
}

// ---------------------------------------------------------------------------
extern "C" void kernel_launch(void* const* d_in, const int* in_sizes, int n_in,
                              void* d_out, int out_size, void* d_ws,
                              size_t ws_size, hipStream_t stream) {
    (void)in_sizes; (void)n_in; (void)out_size; (void)ws_size;
    const int*   words  = (const int*)d_in[0];
    const float* nont   = (const float*)d_in[1];
    const float* root_e = (const float*)d_in[2];
    const float* word_e = (const float*)d_in[3];
    const float* ruleW  = (const float*)d_in[4];
    const float* ruleB  = (const float*)d_in[5];
    // root_mlp params: d_in[6..17]; split_mlp: d_in[18..29]; emit_fc: d_in[30..41]
    const float* eWi  = (const float*)d_in[30]; const float* eBi  = (const float*)d_in[31];
    const float* eW11 = (const float*)d_in[32]; const float* eB11 = (const float*)d_in[33];
    const float* eW12 = (const float*)d_in[34]; const float* eB12 = (const float*)d_in[35];
    const float* eW21 = (const float*)d_in[36]; const float* eB21 = (const float*)d_in[37];
    const float* eW22 = (const float*)d_in[38]; const float* eB22 = (const float*)d_in[39];
    const float* eWo  = (const float*)d_in[40]; const float* eBo  = (const float*)d_in[41];

    char* ws = (char*)d_ws;
    size_t off = 0;
    half_t* wordH  = (half_t*)(ws + off); off += (size_t)512000 * 2;
    half_t* nontH  = (half_t*)(ws + off); off += (size_t)3072 * 2;
    half_t* emitWH = (half_t*)(ws + off); off += (size_t)5 * 16384 * 2;
    half_t* outWH  = (half_t*)(ws + off); off += (size_t)128 * 2;
    float* scores  = (float*)(ws + off); off += (size_t)NROWS * 4;
    float* lseA    = (float*)(ws + off); off += 48 * 4;
    float* rootS   = (float*)(ws + off); off += 48 * 4;
    float* split0  = (float*)(ws + off); off += 48 * 4;
    float* split1  = (float*)(ws + off); off += 48 * 4;
    float* mG      = (float*)(ws + off); off += 48 * 4;
    float* expG    = (float*)(ws + off); off += (size_t)48 * 2304 * 4;
    float* beta    = (float*)(ws + off); off += (size_t)17 * BATCH * LSENT * NTC * 4;

    convert_f16<<<2334, 256, 0, stream>>>(word_e, nont, eWi, eW11, eW12, eW21,
                                          eW22, eWo, wordH, nontH, emitWH, outWH);
    small_mlps<<<1, 64, 0, stream>>>(
        nont, root_e,
        (const float*)d_in[6],  (const float*)d_in[7],  (const float*)d_in[8],
        (const float*)d_in[9],  (const float*)d_in[10], (const float*)d_in[11],
        (const float*)d_in[12], (const float*)d_in[13], (const float*)d_in[14],
        (const float*)d_in[15], (const float*)d_in[16], (const float*)d_in[17],
        (const float*)d_in[18], (const float*)d_in[19], (const float*)d_in[20],
        (const float*)d_in[21], (const float*)d_in[22], (const float*)d_in[23],
        (const float*)d_in[24], (const float*)d_in[25], (const float*)d_in[26],
        (const float*)d_in[27], (const float*)d_in[28], (const float*)d_in[29],
        rootS, split0, split1);
    rule_kernel<<<48, 256, 0, stream>>>(nont, ruleW, ruleB, split0, expG, mG);
    emit_kernel<<<NROWS / 64, 128, 0, stream>>>(wordH, nontH, emitWH, outWH,
                                                eBi, eB11, eB12, eB21, eB22,
                                                eBo, scores);
    lse_kernel<<<48, 256, 0, stream>>>(scores, lseA);
    lex_kernel<<<48, 256, 0, stream>>>(words, scores, lseA, split1, beta);
    for (int w = 2; w <= LSENT; ++w)
        cky_kernel<<<BATCH * (17 - w), 256, 0, stream>>>(w, expG, mG, beta);
    final_kernel<<<1, 64, 0, stream>>>(rootS, beta, (float*)d_out);
}
